// MultiHeadAttention_4569845202994
// MI455X (gfx1250) — matrix operand order repaired and checked by reading, not compiled, not run on hardware
//
#include <hip/hip_runtime.h>
#include <math.h>

#define DEV_INLINE __device__ __forceinline__

typedef __attribute__((ext_vector_type(16))) __bf16 v16bf;
typedef __attribute__((ext_vector_type(8)))  float  v8f;

DEV_INLINE v8f wmma_bf16(v16bf a, v16bf b, v8f c) {
  // D = A(16x32 bf16) * B(32x16 bf16) + C(16x16 f32)
  return __builtin_amdgcn_wmma_f32_16x16x32_bf16(false, a, false, b, (short)0, c, false, false);
}

// Load two 16-byte chunks (8 bf16 each) and assemble one 16-element fragment.
DEV_INLINE v16bf ld2x16(const __bf16* p0, const __bf16* p1) {
  union { uint4 u[2]; v16bf v; } t;
  t.u[0] = *(const uint4*)p0;
  t.u[1] = *(const uint4*)p1;
  return t.v;
}

// ---------------------------------------------------------------------------
// fp32 -> bf16 conversion (vectorized by 4)
// ---------------------------------------------------------------------------
__global__ __launch_bounds__(256) void cvt_bf16(const float* __restrict__ in,
                                                __bf16* __restrict__ out, int n) {
  int i = (blockIdx.x * blockDim.x + threadIdx.x) * 4;
  if (i < n) {
    float4 f = *(const float4*)(in + i);
    out[i + 0] = (__bf16)f.x;
    out[i + 1] = (__bf16)f.y;
    out[i + 2] = (__bf16)f.z;
    out[i + 3] = (__bf16)f.w;
  }
}

// ---------------------------------------------------------------------------
// Batched bf16 GEMM: C[z] = A * B[z] + bias[z]  (block 128x128, K-step 32)
// MODE 0: bf16 out, row-major within z-slab  (Q, K projections)
// MODE 1: bf16 out, V-transposed [b][d][t]   (V projection)
// MODE 2: f32 out + residual                 (final projection)
// ---------------------------------------------------------------------------
template<int MODE>
__global__ __launch_bounds__(256) void gemm_kernel(
    const __bf16* __restrict__ A, const __bf16* __restrict__ Bmat,
    const float* __restrict__ bias, const float* __restrict__ resid,
    void* __restrict__ Out, int M, int N, int K,
    long long bStrideZ, long long biasStrideZ, long long outStrideZ)
{
  __shared__ __align__(16) __bf16 As[128][40];  // [m][k], 8-elem pad keeps 16B align
  __shared__ __align__(16) __bf16 Bs[128][40];  // [n][k] (transposed on store)

  const int tid  = threadIdx.x;
  const int lane = tid & 31;
  const int wid  = tid >> 5;
  const int fr = lane & 15;      // fragment row/col within 16
  const int fg = lane >> 4;      // lane half
  const int wr = (wid >> 1) * 32;  // wave row base (4 waves along M)
  const int wc = (wid & 1) * 64;   // wave col base (2 waves along N)
  const int tileM = blockIdx.y * 128;
  const int tileN = blockIdx.x * 128;
  const int z = blockIdx.z;

  const __bf16* Bz = Bmat + (size_t)z * bStrideZ;
  const float* biasz = bias + (size_t)z * biasStrideZ;
  const long long outOff = (long long)z * outStrideZ;

  const int arow = tid >> 1;        // 0..127
  const int akc  = (tid & 1) * 16;  // 0 / 16
  const int bkr  = tid >> 3;        // 0..31
  const int bnc  = (tid & 7) * 16;  // 0..112

  const v8f vzero = {0.f,0.f,0.f,0.f,0.f,0.f,0.f,0.f};
  v8f acc[2][4];
#pragma unroll
  for (int i = 0; i < 2; ++i)
#pragma unroll
    for (int j = 0; j < 4; ++j) acc[i][j] = vzero;

  for (int k0 = 0; k0 < K; k0 += 32) {
    const __bf16* ap = A + (size_t)(tileM + arow) * K + (k0 + akc);
    uint4 a0 = *(const uint4*)ap;
    uint4 a1 = *(const uint4*)(ap + 8);
    const __bf16* bp = Bz + (size_t)(k0 + bkr) * N + (tileN + bnc);
    union { uint4 u[2]; __bf16 e[16]; } bt;
    bt.u[0] = *(const uint4*)bp;
    bt.u[1] = *(const uint4*)(bp + 8);

    __syncthreads();
    *(uint4*)&As[arow][akc]     = a0;
    *(uint4*)&As[arow][akc + 8] = a1;
#pragma unroll
    for (int j = 0; j < 16; ++j) Bs[bnc + j][bkr] = bt.e[j];
    __syncthreads();

    v16bf af[2], bfrag[4];
#pragma unroll
    for (int ri = 0; ri < 2; ++ri) {
      const __bf16* r0 = &As[wr + ri * 16 + fr][fg * 8];
      const __bf16* r1 = &As[wr + ri * 16 + fr][16 + fg * 8];
      af[ri] = ld2x16(r0, r1);
    }
#pragma unroll
    for (int nj = 0; nj < 4; ++nj) {
      const __bf16* r0 = &Bs[wc + nj * 16 + fr][fg * 8];
      bfrag[nj] = ld2x16(r0, r0 + 16);
    }
#pragma unroll
    for (int ri = 0; ri < 2; ++ri)
#pragma unroll
      for (int nj = 0; nj < 4; ++nj)
        acc[ri][nj] = wmma_bf16(af[ri], bfrag[nj], acc[ri][nj]);
  }

#pragma unroll
  for (int ri = 0; ri < 2; ++ri) {
#pragma unroll
    for (int nj = 0; nj < 4; ++nj) {
      const int n = tileN + wc + nj * 16 + fr;
      const float bi = biasz[n];
#pragma unroll
      for (int r = 0; r < 8; ++r) {
        const int m = tileM + wr + ri * 16 + fg * 8 + r;
        float v = acc[ri][nj][r] + bi;
        if (MODE == 0) {
          ((__bf16*)Out)[outOff + (size_t)m * N + n] = (__bf16)v;
        } else if (MODE == 1) {
          const int bb = m >> 10, tt = m & 1023;         // batch, time
          ((__bf16*)Out)[outOff + ((size_t)bb * 512 + n) * 1024 + tt] = (__bf16)v;
        } else {
          const size_t idx = (size_t)m * N + n;
          ((float*)Out)[idx] = v + resid[idx];
        }
      }
    }
  }
}

// ---------------------------------------------------------------------------
// Flash attention: one wave per (16 query rows) x (128 output dims) x (b,h).
// Q,K row-major bf16 [hb][1024][512]; V transposed bf16 [hb][512][1024].
// Online softmax (fp32), P redistributed C/D->A layout through LDS.
// ---------------------------------------------------------------------------
__global__ __launch_bounds__(32) void attn_kernel(
    const __bf16* __restrict__ Q, const __bf16* __restrict__ Kmat,
    const __bf16* __restrict__ Vt, __bf16* __restrict__ Z, float scale)
{
  __shared__ __align__(16) __bf16 Pl[16][32];
  const int lane = threadIdx.x;
  const int fr = lane & 15, fg = lane >> 4;
  const int i0 = blockIdx.x * 16;    // query row block
  const int d0 = blockIdx.y * 128;   // output-dim split
  const int hb = blockIdx.z;         // h*8 + b

  const __bf16* Qh = Q    + (size_t)hb * (1024 * 512);
  const __bf16* Kh = Kmat + (size_t)hb * (1024 * 512);
  const __bf16* Vh = Vt   + (size_t)hb * (512 * 1024);
  __bf16*       Zh = Z    + (size_t)hb * (1024 * 512);

  const v8f vzero = {0.f,0.f,0.f,0.f,0.f,0.f,0.f,0.f};
  v8f o[8];
  float mrow[8], lrow[8];
#pragma unroll
  for (int j = 0; j < 8; ++j) o[j] = vzero;
#pragma unroll
  for (int r = 0; r < 8; ++r) { mrow[r] = -3.0e38f; lrow[r] = 0.f; }

  const __bf16* qrow = Qh + (size_t)(i0 + fr) * 512;

  for (int sb = 0; sb < 1024; sb += 32) {
    v8f s0 = vzero, s1 = vzero;
    const __bf16* k0p = Kh + (size_t)(sb + fr) * 512 + fg * 16;
    const __bf16* k1p = k0p + 16 * 512;
#pragma unroll 4
    for (int kk = 0; kk < 512; kk += 32) {
      v16bf a  = ld2x16(qrow + kk + fg * 16, qrow + kk + fg * 16 + 8);
      v16bf b0 = ld2x16(k0p + kk, k0p + kk + 8);
      v16bf b1 = ld2x16(k1p + kk, k1p + kk + 8);
      s0 = wmma_bf16(a, b0, s0);
      s1 = wmma_bf16(a, b1, s1);
    }

    float corr[8];
#pragma unroll
    for (int r = 0; r < 8; ++r) {
      float x0 = s0[r] * scale, x1 = s1[r] * scale;
      float mx = fmaxf(x0, x1);
      mx = fmaxf(mx, __shfl_xor(mx, 1, 32));
      mx = fmaxf(mx, __shfl_xor(mx, 2, 32));
      mx = fmaxf(mx, __shfl_xor(mx, 4, 32));
      mx = fmaxf(mx, __shfl_xor(mx, 8, 32));
      float mnew = fmaxf(mrow[r], mx);
      float c  = __expf(mrow[r] - mnew);
      float p0 = __expf(x0 - mnew);
      float p1 = __expf(x1 - mnew);
      float rs = p0 + p1;
      rs += __shfl_xor(rs, 1, 32);
      rs += __shfl_xor(rs, 2, 32);
      rs += __shfl_xor(rs, 4, 32);
      rs += __shfl_xor(rs, 8, 32);
      mrow[r] = mnew;
      lrow[r] = lrow[r] * c + rs;
      corr[r] = c;
      Pl[r + 8 * fg][fr]      = (__bf16)p0;   // C/D layout -> LDS tile [row][s]
      Pl[r + 8 * fg][16 + fr] = (__bf16)p1;
    }
    asm volatile("s_wait_dscnt 0x0" ::: "memory");
    v16bf pa = ld2x16(&Pl[fr][fg * 16], &Pl[fr][fg * 16 + 8]);  // A layout

#pragma unroll
    for (int j = 0; j < 8; ++j)
#pragma unroll
      for (int r = 0; r < 8; ++r) o[j][r] *= corr[r];

#pragma unroll
    for (int j = 0; j < 8; ++j) {
      const __bf16* vp = Vh + (size_t)(d0 + j * 16 + fr) * 1024 + sb + fg * 16;
      v16bf bv = ld2x16(vp, vp + 8);
      o[j] = wmma_bf16(pa, bv, o[j]);
    }
  }

#pragma unroll
  for (int j = 0; j < 8; ++j)
#pragma unroll
    for (int r = 0; r < 8; ++r) {
      float v = o[j][r] / lrow[r];
      Zh[(size_t)(i0 + fg * 8 + r) * 512 + (d0 + j * 16 + fr)] = (__bf16)v;
    }
}

// ---------------------------------------------------------------------------
// Per-head LayerNorm over C=512, writing head-concatenated bf16 [row][h*512+c].
// One wave per (h,row); 16 elems per lane; shuffle reductions.
// ---------------------------------------------------------------------------
__global__ __launch_bounds__(256) void ln_kernel(
    const __bf16* __restrict__ Z, const float* __restrict__ gamma,
    const float* __restrict__ beta, __bf16* __restrict__ Zc)
{
  const int lane = threadIdx.x & 31;
  const int wid  = threadIdx.x >> 5;
  const int gi = blockIdx.x * 8 + wid;   // h*8192 + row
  const int h = gi >> 13;
  const int row = gi & 8191;

  const __bf16* zp = Z + (size_t)gi * 512 + lane * 16;
  union { uint4 u[2]; __bf16 e[16]; } t;
  t.u[0] = *(const uint4*)zp;
  t.u[1] = *(const uint4*)(zp + 8);
  float x[16], sum = 0.f;
#pragma unroll
  for (int j = 0; j < 16; ++j) { x[j] = (float)t.e[j]; sum += x[j]; }
  sum += __shfl_xor(sum, 1, 32);
  sum += __shfl_xor(sum, 2, 32);
  sum += __shfl_xor(sum, 4, 32);
  sum += __shfl_xor(sum, 8, 32);
  sum += __shfl_xor(sum, 16, 32);
  const float mean = sum * (1.0f / 512.0f);
  float sq = 0.f;
#pragma unroll
  for (int j = 0; j < 16; ++j) { float d = x[j] - mean; sq += d * d; }
  sq += __shfl_xor(sq, 1, 32);
  sq += __shfl_xor(sq, 2, 32);
  sq += __shfl_xor(sq, 4, 32);
  sq += __shfl_xor(sq, 8, 32);
  sq += __shfl_xor(sq, 16, 32);
  const float rstd = rsqrtf(sq * (1.0f / 512.0f) + 1e-5f);

  const float* gp = gamma + h * 512 + lane * 16;
  const float* bp = beta  + h * 512 + lane * 16;
  union { uint4 u[2]; __bf16 e[16]; } ot;
#pragma unroll
  for (int j = 0; j < 16; ++j)
    ot.e[j] = (__bf16)((x[j] - mean) * rstd * gp[j] + bp[j]);
  __bf16* op = Zc + (size_t)row * 4096 + h * 512 + lane * 16;
  *(uint4*)op       = ot.u[0];
  *(uint4*)(op + 8) = ot.u[1];
}

// ---------------------------------------------------------------------------
extern "C" void kernel_launch(void* const* d_in, const int* in_sizes, int n_in,
                              void* d_out, int out_size, void* d_ws, size_t ws_size,
                              hipStream_t stream)
{
  const float* x    = (const float*)d_in[0];
  const float* Wq   = (const float*)d_in[1];
  const float* bq   = (const float*)d_in[2];
  const float* Wk   = (const float*)d_in[3];
  const float* bk   = (const float*)d_in[4];
  const float* Wv   = (const float*)d_in[5];
  const float* bv   = (const float*)d_in[6];
  const float* ln_g = (const float*)d_in[7];
  const float* ln_b = (const float*)d_in[8];
  const float* Wf   = (const float*)d_in[9];
  const float* bff  = (const float*)d_in[10];
  float* out = (float*)d_out;

  constexpr int Bb = 8, T = 1024, C = 512, H = 8;
  constexpr int M = Bb * T;           // 8192 tokens
  constexpr int HC = H * C;           // 4096

  char* w = (char*)d_ws;
  size_t off = 0;
  auto wsAlloc = [&](size_t bytes) -> void* {
    void* p = w + off;
    off = (off + bytes + 255) & ~(size_t)255;
    return p;
  };
  __bf16* Xb  = (__bf16*)wsAlloc((size_t)M * C * 2);
  __bf16* Wqb = (__bf16*)wsAlloc((size_t)H * C * C * 2);
  __bf16* Wkb = (__bf16*)wsAlloc((size_t)H * C * C * 2);
  __bf16* Wvb = (__bf16*)wsAlloc((size_t)H * C * C * 2);
  __bf16* Wfb = (__bf16*)wsAlloc((size_t)HC * C * 2);
  __bf16* Qb  = (__bf16*)wsAlloc((size_t)H * M * C * 2);
  __bf16* Kb  = (__bf16*)wsAlloc((size_t)H * M * C * 2);
  __bf16* Vtb = (__bf16*)wsAlloc((size_t)H * M * C * 2);
  __bf16* Zb  = (__bf16*)wsAlloc((size_t)H * M * C * 2);
  __bf16* Zc  = Qb;  // Q is dead after attention; reuse for concat buffer

  auto cvtLaunch = [&](const float* src, __bf16* dst, int n) {
    cvt_bf16<<<dim3((n / 4 + 255) / 256), dim3(256), 0, stream>>>(src, dst, n);
  };
  cvtLaunch(x,  Xb,  M * C);
  cvtLaunch(Wq, Wqb, H * C * C);
  cvtLaunch(Wk, Wkb, H * C * C);
  cvtLaunch(Wv, Wvb, H * C * C);
  cvtLaunch(Wf, Wfb, HC * C);

  dim3 gblk(256);
  dim3 ggrid(C / 128, M / 128, H);   // (4, 64, 8)
  gemm_kernel<0><<<ggrid, gblk, 0, stream>>>(Xb, Wqb, bq, nullptr, Qb,
      M, C, C, (long long)C * C, (long long)C, (long long)M * C);
  gemm_kernel<0><<<ggrid, gblk, 0, stream>>>(Xb, Wkb, bk, nullptr, Kb,
      M, C, C, (long long)C * C, (long long)C, (long long)M * C);
  gemm_kernel<1><<<ggrid, gblk, 0, stream>>>(Xb, Wvb, bv, nullptr, Vtb,
      M, C, C, (long long)C * C, (long long)C, (long long)M * C);

  const float scale = 1.0f / sqrtf((float)C);
  dim3 agrid(T / 16, C / 128, Bb * H);  // (64, 4, 64), 32 threads each
  attn_kernel<<<agrid, dim3(32), 0, stream>>>(Qb, Kb, Vtb, Zb, scale);

  ln_kernel<<<dim3(H * M / 8), dim3(256), 0, stream>>>(Zb, ln_g, ln_b, Zc);

  dim3 fgrid(C / 128, M / 128, 1);
  gemm_kernel<2><<<fgrid, gblk, 0, stream>>>(Zc, Wfb, bff, x, out,
      M, C, HC, 0LL, 0LL, 0LL);
}
